// MPLayer_90503550861444
// MI455X (gfx1250) — matrix operand order repaired and checked by reading, not compiled, not run on hardware
//
#include <hip/hip_runtime.h>
#include <hip/hip_bf16.h>

// ---------------------------------------------------------------------------
// MP layer for gfx1250: edge MLP + scatter-add + node MLP, all GEMMs on
// v_wmma_f32_16x16x32_bf16 (bf16 inputs, fp32 accumulate).
// M=32 per wave: each LDS B-fragment feeds two WMMAs (halves LDS bandwidth
// per FLOP vs the M=16 version, keeping the matrix pipes fed).
// ---------------------------------------------------------------------------

typedef __attribute__((ext_vector_type(16))) __bf16 v16bf;
typedef __attribute__((ext_vector_type(8)))  __bf16 v8bf;
typedef __attribute__((ext_vector_type(8)))  float  v8f;
typedef __attribute__((ext_vector_type(4)))  float  v4f;

#define D_H 128

// ---------------- utility kernels ----------------

__global__ void k_f32_to_bf16(const float* __restrict__ s, __bf16* __restrict__ d, long n) {
  long i = (long)blockIdx.x * blockDim.x + threadIdx.x;
  long stride = (long)gridDim.x * blockDim.x;
  for (; i < n; i += stride) d[i] = (__bf16)s[i];
}

__global__ void k_zero_f32(float* __restrict__ p, long n4) {
  long i = (long)blockIdx.x * blockDim.x + threadIdx.x;
  long stride = (long)gridDim.x * blockDim.x;
  v4f z = {0.f, 0.f, 0.f, 0.f};
  for (; i < n4; i += stride) ((v4f*)p)[i] = z;
}

// Convert fp32 row-major W[K][128] into bf16 fragments in WMMA B layout:
// fragment (kt,nt): lane l holds column N = nt*16 + (l&15),
// elements e<8 hold K = kt*32 + (l>>4)*8 + e, e>=8 hold K = kt*32 + 16 + (l>>4)*8 + (e-8).
__global__ void k_swizzle_w(const float* __restrict__ W, __bf16* __restrict__ out, int K) {
  int total = K * 128;
  int i = blockIdx.x * blockDim.x + threadIdx.x;
  int stride = gridDim.x * blockDim.x;
  for (; i < total; i += stride) {
    int Kk = i >> 7, N = i & 127;
    int kt = Kk >> 5, r = Kk & 31, half = (r >> 3) & 1, e = (r & 7) | ((r >> 4) << 3);
    int nt = N >> 4, l = (half << 4) | (N & 15);
    out[((((kt << 3) + nt) * 32 + l) << 4) + e] = (__bf16)W[i];
  }
}

// ---------------- shared helpers ----------------

static __device__ __forceinline__ v16bf pack16(v8bf lo, v8bf hi) {
  v16bf a;
#pragma unroll
  for (int e = 0; e < 8; ++e) { a[e] = lo[e]; a[e + 8] = hi[e]; }
  return a;
}

static __device__ __forceinline__ float silu(float x) {
  return x / (1.f + __expf(-x));
}

// ---------------- edge message kernel ----------------
// 8 waves / block; each wave: one 32-edge super-tile (two 16-row M tiles),
// full 128-wide output. A = [32 x 256] concat(h[src], h[dst]) gathered
// straight from global bf16; B fragments from LDS-staged swizzled weights,
// each reused by two WMMAs.
__global__ __launch_bounds__(256) void k_edge(
    const __bf16* __restrict__ hbf,
    const int* __restrict__ src, const int* __restrict__ dst,
    const __bf16* __restrict__ W1sw, const __bf16* __restrict__ W2sw,
    const float* __restrict__ b1, const float* __restrict__ b2,
    float* __restrict__ agg, int nE, int nTiles)
{
  __shared__ __bf16 sW1[8 * 8 * 32 * 16];      // 64 KB : W_msg1 (K=256)
  __shared__ __bf16 sW2[4 * 8 * 32 * 16];      // 32 KB : W_msg2 (K=128)
  __shared__ __bf16 sStage[8][32 * D_H];       // 64 KB : per-wave transpose buffer

  for (int i = threadIdx.x; i < 8 * 8 * 32 * 16; i += 256) sW1[i] = W1sw[i];
  for (int i = threadIdx.x; i < 4 * 8 * 32 * 16; i += 256) sW2[i] = W2sw[i];
  __syncthreads();

  const int wave = threadIdx.x >> 5, lane = threadIdx.x & 31;
  const int tile = blockIdx.x * 8 + wave;      // 32-edge super-tile
  if (tile >= nTiles) return;                  // whole-wave exit: EXEC full for WMMA
  const int half = lane >> 4, m = lane & 15;
  const int base = tile * 32;

  int r0 = base + m;      if (r0 >= nE) r0 = nE - 1;
  int r1 = base + 16 + m; if (r1 >= nE) r1 = nE - 1;
  const __bf16* rowS0 = hbf + (long)src[r0] * D_H;
  const __bf16* rowD0 = hbf + (long)dst[r0] * D_H;
  const __bf16* rowS1 = hbf + (long)src[r1] * D_H;
  const __bf16* rowD1 = hbf + (long)dst[r1] * D_H;

  v8f acc0[8], acc1[8];
#pragma unroll
  for (int nt = 0; nt < 8; ++nt)
#pragma unroll
    for (int r = 0; r < 8; ++r) { acc0[nt][r] = 0.f; acc1[nt][r] = 0.f; }

  // ---- GEMM1: [32x256] x [256x128] ----
#pragma unroll
  for (int kt = 0; kt < 8; ++kt) {
    const __bf16* row0 = (kt < 4) ? rowS0 : rowD0;
    const __bf16* row1 = (kt < 4) ? rowS1 : rowD1;
    const int kb = (kt & 3) * 32 + half * 8;   // A layout: two 8-elem runs per lane
    v16bf a0 = pack16(*(const v8bf*)(row0 + kb), *(const v8bf*)(row0 + kb + 16));
    v16bf a1 = pack16(*(const v8bf*)(row1 + kb), *(const v8bf*)(row1 + kb + 16));
#pragma unroll
    for (int nt = 0; nt < 8; ++nt) {
      v16bf b = *(const v16bf*)(sW1 + ((((kt << 3) + nt) * 32 + lane) << 4));
      acc0[nt] = __builtin_amdgcn_wmma_f32_16x16x32_bf16(
          false, a0, false, b, (short)0, acc0[nt], false, false);
      acc1[nt] = __builtin_amdgcn_wmma_f32_16x16x32_bf16(
          false, a1, false, b, (short)0, acc1[nt], false, false);
    }
  }

  // ---- bias + SiLU, transpose C-layout -> A-layout via per-wave LDS ----
  __bf16* st = &sStage[wave][0];
#pragma unroll
  for (int nt = 0; nt < 8; ++nt) {
    const int n = nt * 16 + m;                 // C layout: N = lane&15
    const float bias = b1[n];
#pragma unroll
    for (int r = 0; r < 8; ++r) {              // C layout: M = r + 8*(lane>>4)
      st[(r + 8 * half) * D_H + n]        = (__bf16)silu(acc0[nt][r] + bias);
      st[(16 + r + 8 * half) * D_H + n]   = (__bf16)silu(acc1[nt][r] + bias);
    }
  }

  // ---- GEMM2: [32x128] x [128x128] ----
  v8f acc20[8], acc21[8];
#pragma unroll
  for (int nt = 0; nt < 8; ++nt)
#pragma unroll
    for (int r = 0; r < 8; ++r) { acc20[nt][r] = 0.f; acc21[nt][r] = 0.f; }

  const __bf16* arow0 = st + m * D_H;
  const __bf16* arow1 = st + (16 + m) * D_H;
#pragma unroll
  for (int kt = 0; kt < 4; ++kt) {
    const int kb = kt * 32 + half * 8;
    v16bf a0 = pack16(*(const v8bf*)(arow0 + kb), *(const v8bf*)(arow0 + kb + 16));
    v16bf a1 = pack16(*(const v8bf*)(arow1 + kb), *(const v8bf*)(arow1 + kb + 16));
#pragma unroll
    for (int nt = 0; nt < 8; ++nt) {
      v16bf b = *(const v16bf*)(sW2 + ((((kt << 3) + nt) * 32 + lane) << 4));
      acc20[nt] = __builtin_amdgcn_wmma_f32_16x16x32_bf16(
          false, a0, false, b, (short)0, acc20[nt], false, false);
      acc21[nt] = __builtin_amdgcn_wmma_f32_16x16x32_bf16(
          false, a1, false, b, (short)0, acc21[nt], false, false);
    }
  }

  // ---- + b2, scatter-add into agg[dst] (resolves in L2) ----
  int dn0[8], dn1[8];
#pragma unroll
  for (int r = 0; r < 8; ++r) {
    int e0 = base + r + 8 * half;      if (e0 >= nE) e0 = nE - 1;
    int e1 = base + 16 + r + 8 * half; if (e1 >= nE) e1 = nE - 1;
    dn0[r] = dst[e0];
    dn1[r] = dst[e1];
  }
#pragma unroll
  for (int nt = 0; nt < 8; ++nt) {
    const int n = nt * 16 + m;
    const float bias = b2[n];
#pragma unroll
    for (int r = 0; r < 8; ++r) {
      __hip_atomic_fetch_add(&agg[(long)dn0[r] * D_H + n], acc20[nt][r] + bias,
                             __ATOMIC_RELAXED, __HIP_MEMORY_SCOPE_AGENT);
      __hip_atomic_fetch_add(&agg[(long)dn1[r] * D_H + n], acc21[nt][r] + bias,
                             __ATOMIC_RELAXED, __HIP_MEMORY_SCOPE_AGENT);
    }
  }
}

// ---------------- node update kernel ----------------
__global__ __launch_bounds__(256) void k_node(
    const __bf16* __restrict__ hbf, const float* __restrict__ aggf,
    const __bf16* __restrict__ W1sw, const __bf16* __restrict__ W2sw,
    const float* __restrict__ b1, const float* __restrict__ b2,
    float* __restrict__ out, int nN, int nTiles)
{
  __shared__ __bf16 sW1[8 * 8 * 32 * 16];      // 64 KB : W_upd1
  __shared__ __bf16 sW2[4 * 8 * 32 * 16];      // 32 KB : W_upd2
  __shared__ __bf16 sStage[8][32 * D_H];       // 64 KB

  for (int i = threadIdx.x; i < 8 * 8 * 32 * 16; i += 256) sW1[i] = W1sw[i];
  for (int i = threadIdx.x; i < 4 * 8 * 32 * 16; i += 256) sW2[i] = W2sw[i];
  __syncthreads();

  const int wave = threadIdx.x >> 5, lane = threadIdx.x & 31;
  const int tile = blockIdx.x * 8 + wave;      // 32-node super-tile
  if (tile >= nTiles) return;
  const int half = lane >> 4, m = lane & 15;
  const int base = tile * 32;

  int n0 = base + m;      if (n0 >= nN) n0 = nN - 1;
  int n1 = base + 16 + m; if (n1 >= nN) n1 = nN - 1;
  const __bf16* rowH0 = hbf + (long)n0 * D_H;
  const __bf16* rowH1 = hbf + (long)n1 * D_H;
  const float*  rowA0 = aggf + (long)n0 * D_H;
  const float*  rowA1 = aggf + (long)n1 * D_H;

  v8f acc0[8], acc1[8];
#pragma unroll
  for (int nt = 0; nt < 8; ++nt)
#pragma unroll
    for (int r = 0; r < 8; ++r) { acc0[nt][r] = 0.f; acc1[nt][r] = 0.f; }

  // ---- GEMM1: [32x256] (concat(h, agg)) x [256x128] ----
#pragma unroll
  for (int kt = 0; kt < 8; ++kt) {
    v16bf a0, a1;
    if (kt < 4) {
      const int kb = kt * 32 + half * 8;
      a0 = pack16(*(const v8bf*)(rowH0 + kb), *(const v8bf*)(rowH0 + kb + 16));
      a1 = pack16(*(const v8bf*)(rowH1 + kb), *(const v8bf*)(rowH1 + kb + 16));
    } else {
      const int kb = (kt - 4) * 32 + half * 8;
      v4f p0 = *(const v4f*)(rowA0 + kb),      p1 = *(const v4f*)(rowA0 + kb + 4);
      v4f p2 = *(const v4f*)(rowA0 + kb + 16), p3 = *(const v4f*)(rowA0 + kb + 20);
      v4f q0 = *(const v4f*)(rowA1 + kb),      q1 = *(const v4f*)(rowA1 + kb + 4);
      v4f q2 = *(const v4f*)(rowA1 + kb + 16), q3 = *(const v4f*)(rowA1 + kb + 20);
#pragma unroll
      for (int e = 0; e < 4; ++e) {
        a0[e] = (__bf16)p0[e]; a0[e + 4]  = (__bf16)p1[e];
        a0[e + 8] = (__bf16)p2[e]; a0[e + 12] = (__bf16)p3[e];
        a1[e] = (__bf16)q0[e]; a1[e + 4]  = (__bf16)q1[e];
        a1[e + 8] = (__bf16)q2[e]; a1[e + 12] = (__bf16)q3[e];
      }
    }
#pragma unroll
    for (int nt = 0; nt < 8; ++nt) {
      v16bf b = *(const v16bf*)(sW1 + ((((kt << 3) + nt) * 32 + lane) << 4));
      acc0[nt] = __builtin_amdgcn_wmma_f32_16x16x32_bf16(
          false, a0, false, b, (short)0, acc0[nt], false, false);
      acc1[nt] = __builtin_amdgcn_wmma_f32_16x16x32_bf16(
          false, a1, false, b, (short)0, acc1[nt], false, false);
    }
  }

  // ---- bias + SiLU + transpose ----
  __bf16* st = &sStage[wave][0];
#pragma unroll
  for (int nt = 0; nt < 8; ++nt) {
    const int n = nt * 16 + m;
    const float bias = b1[n];
#pragma unroll
    for (int r = 0; r < 8; ++r) {
      st[(r + 8 * half) * D_H + n]      = (__bf16)silu(acc0[nt][r] + bias);
      st[(16 + r + 8 * half) * D_H + n] = (__bf16)silu(acc1[nt][r] + bias);
    }
  }

  // ---- GEMM2: [32x128] x [128x128] ----
  v8f acc20[8], acc21[8];
#pragma unroll
  for (int nt = 0; nt < 8; ++nt)
#pragma unroll
    for (int r = 0; r < 8; ++r) { acc20[nt][r] = 0.f; acc21[nt][r] = 0.f; }

  const __bf16* arow0 = st + m * D_H;
  const __bf16* arow1 = st + (16 + m) * D_H;
#pragma unroll
  for (int kt = 0; kt < 4; ++kt) {
    const int kb = kt * 32 + half * 8;
    v16bf a0 = pack16(*(const v8bf*)(arow0 + kb), *(const v8bf*)(arow0 + kb + 16));
    v16bf a1 = pack16(*(const v8bf*)(arow1 + kb), *(const v8bf*)(arow1 + kb + 16));
#pragma unroll
    for (int nt = 0; nt < 8; ++nt) {
      v16bf b = *(const v16bf*)(sW2 + ((((kt << 3) + nt) * 32 + lane) << 4));
      acc20[nt] = __builtin_amdgcn_wmma_f32_16x16x32_bf16(
          false, a0, false, b, (short)0, acc20[nt], false, false);
      acc21[nt] = __builtin_amdgcn_wmma_f32_16x16x32_bf16(
          false, a1, false, b, (short)0, acc21[nt], false, false);
    }
  }

  // ---- + b2, write output fp32 ----
#pragma unroll
  for (int nt = 0; nt < 8; ++nt) {
    const int n = nt * 16 + m;
    const float bias = b2[n];
#pragma unroll
    for (int r = 0; r < 8; ++r) {
      int o0 = base + r + 8 * half;
      int o1 = base + 16 + r + 8 * half;
      if (o0 < nN) out[(long)o0 * D_H + n] = acc20[nt][r] + bias;
      if (o1 < nN) out[(long)o1 * D_H + n] = acc21[nt][r] + bias;
    }
  }
}

// ---------------- host entry ----------------
extern "C" void kernel_launch(void* const* d_in, const int* in_sizes, int n_in,
                              void* d_out, int out_size, void* d_ws, size_t ws_size,
                              hipStream_t stream)
{
  const float* h   = (const float*)d_in[0];
  const int*   ei  = (const int*)d_in[1];
  const float* Wm1 = (const float*)d_in[2];
  const float* bm1 = (const float*)d_in[3];
  const float* Wm2 = (const float*)d_in[4];
  const float* bm2 = (const float*)d_in[5];
  const float* Wu1 = (const float*)d_in[6];
  const float* bu1 = (const float*)d_in[7];
  const float* Wu2 = (const float*)d_in[8];
  const float* bu2 = (const float*)d_in[9];

  const int nN = in_sizes[0] / D_H;   // 50000
  const int nE = in_sizes[1] / 2;     // 800000
  const int* src = ei;
  const int* dst = ei + nE;

  // carve workspace (~38.6 MB)
  char* ws = (char*)d_ws;
  size_t off = 0;
  auto carve = [&](size_t bytes) -> void* {
    void* p = ws + off;
    off = (off + bytes + 255) & ~(size_t)255;
    return p;
  };
  __bf16* hbf = (__bf16*)carve((size_t)nN * D_H * 2);
  __bf16* W1m = (__bf16*)carve(256 * 128 * 2);
  __bf16* W2m = (__bf16*)carve(128 * 128 * 2);
  __bf16* W1u = (__bf16*)carve(256 * 128 * 2);
  __bf16* W2u = (__bf16*)carve(128 * 128 * 2);
  float*  agg = (float*)carve((size_t)nN * D_H * 4);
  (void)ws_size; (void)n_in; (void)out_size;

  const long nH = (long)nN * D_H;
  k_f32_to_bf16<<<1024, 256, 0, stream>>>(h, hbf, nH);
  k_swizzle_w<<<128, 256, 0, stream>>>(Wm1, W1m, 256);
  k_swizzle_w<<<64,  256, 0, stream>>>(Wm2, W2m, 128);
  k_swizzle_w<<<128, 256, 0, stream>>>(Wu1, W1u, 256);
  k_swizzle_w<<<64,  256, 0, stream>>>(Wu2, W2u, 128);
  k_zero_f32<<<1024, 256, 0, stream>>>(agg, nH / 4);

  const int tilesE = (nE + 31) / 32;  // 32-edge super-tiles
  const int tilesN = (nN + 31) / 32;  // 32-node super-tiles
  k_edge<<<(tilesE + 7) / 8, 256, 0, stream>>>(hbf, src, dst, W1m, W2m, bm1, bm2,
                                               agg, nE, tilesE);
  k_node<<<(tilesN + 7) / 8, 256, 0, stream>>>(hbf, agg, W1u, W2u, bu1, bu2,
                                               (float*)d_out, nN, tilesN);
}
